// GeoNetDepthToNormal_76244259438620
// MI455X (gfx1250) — matrix operand order repaired and checked by reading, not compiled, not run on hardware
//
#include <hip/hip_runtime.h>
#include <hip/hip_bf16.h>
#include <cstdint>

// ---------------------------------------------------------------------------
// GeoNet depth->normal pipeline for MI455X (gfx1250, wave32, WMMA).
// conv1/conv2 are implicit-GEMM via v_wmma_f32_16x16x32_f16.
// conv2: 64-pixel row tile staged in LDS (halo, f16), 4 M-tiles per wave
// reusing one B fragment -> 72 WMMAs/wave, A-operand served from ds_load_b128.
// Staging uses global_load_async_to_lds_b128 (probe-confirmed to exist here;
// param1 is int4 AS1*) with s_wait_asynccnt before the barrier.
// ---------------------------------------------------------------------------

typedef __attribute__((ext_vector_type(16))) _Float16 v16h;
typedef __attribute__((ext_vector_type(8)))  _Float16 v8h;
typedef __attribute__((ext_vector_type(4)))  _Float16 v4h;
typedef __attribute__((ext_vector_type(8)))  float    v8f;
typedef __attribute__((ext_vector_type(4)))  int      v4i;

#define BATCH 4
#define HH 256
#define WW 256
#define HWPIX 65536          // H*W
#define NPIX  262144         // B*H*W
#define GAMMA_TH 0.95f
#define EPS_V 1e-5f
#define MT 4                 // M-tiles (of 16 pixels) per block

#if __has_builtin(__builtin_amdgcn_global_load_async_to_lds_b128) && \
    __has_builtin(__builtin_amdgcn_s_wait_asynccnt)
#define USE_ASYNC_LDS 1
// address-space-qualified pointee typedefs for the async builtin
typedef __attribute__((address_space(1))) v4i gas_v4i;   // global
typedef __attribute__((address_space(3))) v4i las_v4i;   // LDS
#else
#define USE_ASYNC_LDS 0
#endif

// ===========================================================================
// 1) depth -> normal_from_depth (9x9 least-squares plane fit, edge padded)
// ===========================================================================
__global__ __launch_bounds__(256)
void k_normal_from_depth(const float* __restrict__ depth,
                         const float* __restrict__ fov,
                         float* __restrict__ nfd)
{
    const int b  = blockIdx.z;
    const int h0 = blockIdx.y * 16;
    const int w0 = blockIdx.x * 16;

    __shared__ float sz[24][25];      // (16+2*4)^2 depth tile, padded stride
    const int tid = threadIdx.y * 16 + threadIdx.x;
    const float* dptr = depth + (size_t)b * HWPIX;

    for (int i = tid; i < 24 * 24; i += 256) {
        int r = i / 24, c = i - r * 24;
        int yy = min(max(h0 - 4 + r, 0), HH - 1);
        int xx = min(max(w0 - 4 + c, 0), WW - 1);
        float z = dptr[yy * WW + xx];
        sz[r][c] = (z < EPS_V) ? EPS_V : z;
    }
    __syncthreads();

    const int h = h0 + threadIdx.y;
    const int w = w0 + threadIdx.x;

    const float t    = 2.0f * tanf(0.5f * fov[b]);
    const float finv = t / 256.0f;          // 1/fx == 1/fy (H==W)
    const float cx   = 127.5f;

    const float cz = sz[threadIdx.y + 4][threadIdx.x + 4];

    float a00=0.f,a01=0.f,a02=0.f,a11=0.f,a12=0.f,a22=0.f;
    float t0=0.f,t1=0.f,t2=0.f;

    for (int dy = -4; dy <= 4; ++dy) {
        for (int dx = -4; dx <= 4; ++dx) {
            int yy = min(max(h + dy, 0), HH - 1);
            int xx = min(max(w + dx, 0), WW - 1);
            float z  = sz[threadIdx.y + 4 + dy][threadIdx.x + 4 + dx];
            float zd = (z - cz) / cz;
            if (fabsf(zd) > GAMMA_TH) continue;   // masked tap -> p = 0
            float px = z * ((float)xx - cx) * finv;
            float py = z * ((float)yy - cx) * finv;
            a00 += px*px; a01 += px*py; a02 += px*z;
            a11 += py*py; a12 += py*z;  a22 += z*z;
            t0 += px; t1 += py; t2 += z;
        }
    }

    float det = a00*(a11*a22 - a12*a12)
              - a01*(a01*a22 - a12*a02)
              + a02*(a01*a12 - a11*a02);

    float n0, n1, n2;
    if (det < EPS_V) {                      // A <- I  => x = Atb
        n0 = t0; n1 = t1; n2 = t2;
    } else {
        float i00 = (a11*a22 - a12*a12), i01 = (a02*a12 - a01*a22);
        float i02 = (a01*a12 - a02*a11), i11 = (a00*a22 - a02*a02);
        float i12 = (a01*a02 - a00*a12), i22 = (a00*a11 - a01*a01);
        float r = 1.0f / det;
        n0 = (i00*t0 + i01*t1 + i02*t2) * r;
        n1 = (i01*t0 + i11*t1 + i12*t2) * r;
        n2 = (i02*t0 + i12*t1 + i22*t2) * r;
    }
    float nn  = sqrtf(n0*n0 + n1*n1 + n2*n2);
    float inv = (nn > 0.f) ? (1.0f / nn) : 0.0f;    // NaN -> 0 guard
    if (!(inv == inv)) inv = 0.0f;
    n0 = -n0 * inv; n1 = -n1 * inv; n2 = -n2 * inv;

    const int hw = h * WW + w;
    nfd[((b*3 + 0) << 16) + hw] = n0;
    nfd[((b*3 + 1) << 16) + hw] = n1;
    nfd[((b*3 + 2) << 16) + hw] = n2;
}

// ===========================================================================
// 2) Weight packing into WMMA B-fragment lane order.
//    B fragment (K=32 x N=16, f16): lane half h, element j holds
//    K = 8h+j (j<8) else 16+8h+(j-8); column N = lane%16.
// ===========================================================================
__global__ void k_pack_w1(const float* __restrict__ w1, _Float16* __restrict__ wpk)
{
    // layout: [nt(4)][lane(32)][j(16)] ; K = im2col index (tap*3 + cin), <27
    int idx = blockIdx.x * 256 + threadIdx.x;        // 4*32*16 = 2048
    if (idx >= 4 * 32 * 16) return;
    int j    = idx & 15;
    int lane = (idx >> 4) & 31;
    int nt   = idx >> 9;
    int co   = nt * 16 + (lane & 15);
    int K    = (j < 8) ? (((lane >> 4) << 3) + j) : (16 + ((lane >> 4) << 3) + (j - 8));  // 0..31
    _Float16 v = (_Float16)0.0f;
    if (K < 27) {
        int tt = K / 3, c = K - tt * 3;
        int dh = tt / 3, dw = tt - dh * 3;
        v = (_Float16)w1[((co * 3 + c) * 3 + dh) * 3 + dw];   // OIHW (64,3,3,3)
    }
    wpk[idx] = v;
}

__global__ void k_pack_w2(const float* __restrict__ w2, _Float16* __restrict__ wpk)
{
    // layout: [tap(9)][kc(2)][nt(4)][lane(32)][j(16)]
    int idx = blockIdx.x * 256 + threadIdx.x;        // 9*2*4*32*16 = 36864
    if (idx >= 9 * 2 * 4 * 32 * 16) return;
    int j    = idx & 15;
    int lane = (idx >> 4) & 31;
    int nt   = (idx >> 9) & 3;
    int kc   = (idx >> 11) & 1;
    int tap  = idx >> 12;
    int co   = nt * 16 + (lane & 15);
    int cin  = kc * 32 + ((j < 8) ? (((lane >> 4) << 3) + j) : (16 + ((lane >> 4) << 3) + (j - 8)));     // K within 64 in-channels
    int dh   = tap / 3, dw = tap - dh * 3;
    wpk[idx] = (_Float16)w2[((co * 64 + cin) * 3 + dh) * 3 + dw];  // OIHW (64,64,3,3)
}

// ===========================================================================
// 3) conv1: 3->64, 3x3, zero pad, im2col K=27 padded to one 32-chunk.
//    Block = 128 thr = 4 waves; wave nt -> out-channels nt*16..+15.
//    MT=4 M-tiles per wave reuse one B fragment (4 WMMAs).
// ===========================================================================
__global__ __launch_bounds__(128)
void k_conv1_wmma(const float* __restrict__ nfd,
                  const _Float16* __restrict__ wpk,
                  const float* __restrict__ bias,
                  float* __restrict__ out)
{
    const int lane = threadIdx.x & 31;
    const int nt   = threadIdx.x >> 5;
    const int m    = lane & 15;
    const int p0   = blockIdx.x * (16 * MT);  // 64 pixels, one row, one batch
    const int b    = p0 >> 16;
    const int h    = (p0 & 65535) >> 8;
    const int w0   = p0 & 255;
    const int base = (lane < 16) ? 0 : 8;     // A-fragment K-half select

    // B fragment: load once, reuse across MT tiles
    union { v16h v; v8h h8[2]; } Bf;
    const v8h* wb = reinterpret_cast<const v8h*>(wpk + ((nt * 32 + lane) << 4));
    Bf.h8[0] = wb[0];
    Bf.h8[1] = wb[1];

    const float* src = nfd + (size_t)b * 3 * HWPIX;
    v8f acc[MT];

    #pragma unroll
    for (int mt = 0; mt < MT; ++mt) {
        const int w = w0 + mt * 16 + m;
        union { v16h v; _Float16 e[16]; } A;
        #pragma unroll
        for (int j = 0; j < 16; ++j) {
            // A lane layout: K = base+j (j<8) else 16+base+(j-8)
            int K = (j < 8) ? (base + j) : (16 + base + (j - 8));
            _Float16 val = (_Float16)0.0f;
            if (K < 27) {
                int tt = K / 3, c = K - tt * 3;
                int dy = tt / 3 - 1, dx = tt % 3 - 1;
                int yy = h + dy, xx = w + dx;
                if (yy >= 0 && yy < HH && xx >= 0 && xx < WW)
                    val = (_Float16)(10.0f * src[(c << 16) + (yy << 8) + xx]);
            }
            A.e[j] = val;
        }
        v8f c0 = {};
        acc[mt] = __builtin_amdgcn_wmma_f32_16x16x32_f16(false, A.v, false, Bf.v,
                                                         (short)0, c0, false, false);
    }

    const int co = nt * 16 + m;
    const float bs = bias[co];
    #pragma unroll
    for (int mt = 0; mt < MT; ++mt) {
        #pragma unroll
        for (int r = 0; r < 8; ++r) {
            int mm = r + ((lane < 16) ? 0 : 8);   // D: VGPR r -> row r / 8+r
            int q  = p0 + mt * 16 + mm;
            float v = acc[mt][r] + bs;
            out[(size_t)q * 64 + co] = (v > 0.0f) ? v : 0.0f;
        }
    }
}

// ===========================================================================
// 4) conv2: 64->64, 3x3, implicit GEMM.
//    Block = 128 thr = 4 waves, 64 pixels of one row. Activation halo tile
//    (3 rows x 66 cols x 64 ch f16 = 25 KB) staged in LDS once, A-fragments
//    from ds_load_b128, one B fragment per (tap,kc) reused by 4 M-tiles:
//    9*2*4 = 72 WMMAs per wave.
// ===========================================================================
__global__ __launch_bounds__(128)
void k_conv2_wmma(const _Float16* __restrict__ act,
                  const _Float16* __restrict__ wpk,
                  const float* __restrict__ bias,
                  float* __restrict__ out)
{
    const int tid  = threadIdx.x;
    const int lane = tid & 31;
    const int nt   = tid >> 5;
    const int m    = lane & 15;
    const int p0   = blockIdx.x * (16 * MT);  // 64 pixels, one row, one batch
    const int b    = p0 >> 16;
    const int h    = (p0 & 65535) >> 8;
    const int w0   = p0 & 255;                // multiple of 64
    const int base = (lane < 16) ? 0 : 8;

    __shared__ _Float16 tile[3 * 66 * 64];    // [row 0..2][col 0..65][ch]

    // ---- zero-fill (covers halo / OOB), then stage in-bounds pixels -------
    {
        v8h z8 = {};
        v8h* tv = reinterpret_cast<v8h*>(tile);
        for (int i = tid; i < (3 * 66 * 64) / 8; i += 128) tv[i] = z8;
    }
    __syncthreads();

    for (int s = tid; s < 3 * 66; s += 128) {
        int r = s / 66, c = s - r * 66;
        int yy = h + r - 1;
        int xx = w0 + c - 1;
        if (yy >= 0 && yy < HH && xx >= 0 && xx < WW) {
            const _Float16* gp = act + ((size_t)((b << 16) + (yy << 8) + xx) << 6);
            _Float16* lp = tile + ((r * 66 + c) << 6);
            #pragma unroll
            for (int k = 0; k < 8; ++k) {
#if USE_ASYNC_LDS
                __builtin_amdgcn_global_load_async_to_lds_b128(
                    (gas_v4i*)(uintptr_t)(gp + k * 8),
                    (las_v4i*)(uintptr_t)(lp + k * 8),
                    0, 0);
#else
                *reinterpret_cast<v8h*>(lp + k * 8) =
                    *reinterpret_cast<const v8h*>(gp + k * 8);
#endif
            }
        }
    }
#if USE_ASYNC_LDS
    __builtin_amdgcn_s_wait_asynccnt(0);
#endif
    __syncthreads();

    // ---- GEMM loop --------------------------------------------------------
    v8f acc[MT] = {};
    union { v16h v; v8h h8[2]; } A, Bf;

    #pragma unroll
    for (int t = 0; t < 9; ++t) {
        const int dy = t / 3 - 1, dx = t % 3 - 1;
        if (t < 8)   // warm GL2 for the next tap's packed weights
            __builtin_prefetch(wpk + ((((t + 1) * 2 * 4 + nt) * 32 + lane) << 4), 0, 0);

        #pragma unroll
        for (int kc = 0; kc < 2; ++kc) {
            const v8h* wb = reinterpret_cast<const v8h*>(
                wpk + (((((t * 2 + kc) * 4) + nt) * 32 + lane) << 4));
            Bf.h8[0] = wb[0];
            Bf.h8[1] = wb[1];

            #pragma unroll
            for (int mt = 0; mt < MT; ++mt) {
                const int col = mt * 16 + m + dx + 1;      // 0..65
                const int row = dy + 1;                    // 0..2
                const _Float16* lp = tile + ((row * 66 + col) << 6) + kc * 32 + base;
                const v8h* a8 = reinterpret_cast<const v8h*>(lp);
                A.h8[0] = a8[0];      // K = cbase .. cbase+7
                A.h8[1] = a8[2];      // K = cbase+16 .. cbase+23
                acc[mt] = __builtin_amdgcn_wmma_f32_16x16x32_f16(
                    false, A.v, false, Bf.v, (short)0, acc[mt], false, false);
            }
        }
    }

    // ---- epilogue: bias + relu, NHWC f32 ---------------------------------
    const int co = nt * 16 + m;
    const float bs = bias[co];
    #pragma unroll
    for (int mt = 0; mt < MT; ++mt) {
        #pragma unroll
        for (int r = 0; r < 8; ++r) {
            int mm = r + ((lane < 16) ? 0 : 8);
            int q  = p0 + mt * 16 + mm;
            float v = acc[mt][r] + bs;
            out[(size_t)q * 64 + co] = (v > 0.0f) ? v : 0.0f;
        }
    }
}

// ===========================================================================
// 5) GroupNorm: stats (block per (b,group), LDS tree reduce) + fused
//    apply/convert-to-f16 (vectorized: float4 in, 4 x f16 out).
// ===========================================================================
__global__ __launch_bounds__(256)
void k_gn_stats(const float* __restrict__ x, float* __restrict__ stats)
{
    const int b = blockIdx.x >> 3;
    const int g = blockIdx.x & 7;
    const int tid = threadIdx.x;
    float s = 0.f, ss = 0.f;
    const float* bp = x + (size_t)b * HWPIX * 64 + g * 8;
    for (int i = tid; i < HWPIX; i += 256) {
        const float* pp = bp + (size_t)i * 64;
        #pragma unroll
        for (int c = 0; c < 8; ++c) { float v = pp[c]; s += v; ss += v * v; }
    }
    __shared__ float sh_s[256], sh_q[256];
    sh_s[tid] = s; sh_q[tid] = ss;
    __syncthreads();
    for (int off = 128; off > 0; off >>= 1) {
        if (tid < off) { sh_s[tid] += sh_s[tid + off]; sh_q[tid] += sh_q[tid + off]; }
        __syncthreads();
    }
    if (tid == 0) {
        const float n = (float)HWPIX * 8.0f;
        float mean = sh_s[0] / n;
        float var  = sh_q[0] / n - mean * mean;
        stats[blockIdx.x * 2 + 0] = mean;
        stats[blockIdx.x * 2 + 1] = rsqrtf(var + 1e-5f);
    }
}

__global__ __launch_bounds__(256)
void k_gn_apply_pack(const float* __restrict__ x, const float* __restrict__ stats,
                     const float* __restrict__ gamma, const float* __restrict__ beta,
                     _Float16* __restrict__ y)
{
    // one thread = 4 consecutive channels (same group since 4 | 8)
    size_t idx = (size_t)blockIdx.x * 256 + threadIdx.x;   // NPIX*16 threads
    int c4 = (int)(idx & 15) * 4;
    size_t q = idx >> 4;
    int b = (int)(q >> 16);
    int g = c4 >> 3;
    float mean = stats[(b * 8 + g) * 2 + 0];
    float rstd = stats[(b * 8 + g) * 2 + 1];
    const float4 xv = *reinterpret_cast<const float4*>(x + q * 64 + c4);
    v4h o;
    o[0] = (_Float16)((xv.x - mean) * rstd * gamma[c4 + 0] + beta[c4 + 0]);
    o[1] = (_Float16)((xv.y - mean) * rstd * gamma[c4 + 1] + beta[c4 + 1]);
    o[2] = (_Float16)((xv.z - mean) * rstd * gamma[c4 + 2] + beta[c4 + 2]);
    o[3] = (_Float16)((xv.w - mean) * rstd * gamma[c4 + 3] + beta[c4 + 3]);
    *reinterpret_cast<v4h*>(y + q * 64 + c4) = o;
}

// ===========================================================================
// 6) conv3: 64->3, 3x3 (no relu). 3 outputs -> VALU with weights in LDS.
// ===========================================================================
__global__ __launch_bounds__(256)
void k_conv3(const _Float16* __restrict__ act, const float* __restrict__ w3,
             const float* __restrict__ b3, float* __restrict__ res)
{
    __shared__ float sw[3 * 64 * 9];
    for (int i = threadIdx.x; i < 3 * 64 * 9; i += 256) sw[i] = w3[i];
    __syncthreads();

    const int p  = blockIdx.x * 256 + threadIdx.x;
    const int b  = p >> 16;
    const int hw = p & 65535;
    const int h  = hw >> 8;
    const int w  = hw & 255;

    float a0 = b3[0], a1 = b3[1], a2 = b3[2];
    for (int t = 0; t < 9; ++t) {
        int dy = t / 3 - 1, dx = t % 3 - 1;
        int yy = h + dy, xx = w + dx;
        if (yy < 0 || yy >= HH || xx < 0 || xx >= WW) continue;
        const _Float16* ap = act + ((size_t)((b << 16) + (yy << 8) + xx) << 6);
        #pragma unroll 8
        for (int cin = 0; cin < 64; ++cin) {
            float v = (float)ap[cin];
            a0 += v * sw[(0 * 64 + cin) * 9 + t];
            a1 += v * sw[(1 * 64 + cin) * 9 + t];
            a2 += v * sw[(2 * 64 + cin) * 9 + t];
        }
    }
    res[((b * 3 + 0) << 16) + hw] = a0;
    res[((b * 3 + 1) << 16) + hw] = a1;
    res[((b * 3 + 2) << 16) + hw] = a2;
}

// ===========================================================================
// 7) Fused tail: s = normalize(nfd + res3); cat(normals, s); 1x1 conv 6->3;
//    normalize; (x+1)/2; clip.
// ===========================================================================
__global__ __launch_bounds__(256)
void k_final(const float* __restrict__ normals, const float* __restrict__ nfd,
             const float* __restrict__ res3, const float* __restrict__ wfc,
             const float* __restrict__ bfc, float* __restrict__ out)
{
    const int p  = blockIdx.x * 256 + threadIdx.x;
    const int b  = p >> 16;
    const int hw = p & 65535;
    const float* nb = normals + (size_t)b * 3 * HWPIX;
    const float* fb = nfd     + (size_t)b * 3 * HWPIX;
    const float* rb = res3    + (size_t)b * 3 * HWPIX;

    float s0 = fb[hw]             + rb[hw];
    float s1 = fb[HWPIX + hw]     + rb[HWPIX + hw];
    float s2 = fb[2 * HWPIX + hw] + rb[2 * HWPIX + hw];
    float nn = sqrtf(s0 * s0 + s1 * s1 + s2 * s2);
    s0 /= nn; s1 /= nn; s2 /= nn;

    float cat[6];
    cat[0] = nb[hw]; cat[1] = nb[HWPIX + hw]; cat[2] = nb[2 * HWPIX + hw];
    cat[3] = s0;     cat[4] = s1;             cat[5] = s2;

    float o[3];
    #pragma unroll
    for (int co = 0; co < 3; ++co) {
        float a = bfc[co];
        #pragma unroll
        for (int k = 0; k < 6; ++k) a += wfc[co * 6 + k] * cat[k];
        o[co] = a;
    }
    float on = sqrtf(o[0]*o[0] + o[1]*o[1] + o[2]*o[2]);
    #pragma unroll
    for (int co = 0; co < 3; ++co) {
        float v = (o[co] / on + 1.0f) * 0.5f;
        v = fminf(fmaxf(v, 0.0f), 1.0f);
        out[((b * 3 + co) << 16) + hw] = v;
    }
}

// ===========================================================================
// Host launcher
// ===========================================================================
extern "C" void kernel_launch(void* const* d_in, const int* in_sizes, int n_in,
                              void* d_out, int out_size, void* d_ws, size_t ws_size,
                              hipStream_t stream) {
    const float* initial_depth   = (const float*)d_in[0];
    const float* x_fov           = (const float*)d_in[1];
    const float* initial_normals = (const float*)d_in[2];
    const float* w1  = (const float*)d_in[3];
    const float* b1  = (const float*)d_in[4];
    const float* g1  = (const float*)d_in[5];
    const float* bt1 = (const float*)d_in[6];
    const float* w2  = (const float*)d_in[7];
    const float* b2  = (const float*)d_in[8];
    const float* g2  = (const float*)d_in[9];
    const float* bt2 = (const float*)d_in[10];
    const float* w3  = (const float*)d_in[11];
    const float* b3  = (const float*)d_in[12];
    const float* wfc = (const float*)d_in[13];
    const float* bfc = (const float*)d_in[14];
    float* out = (float*)d_out;

    // workspace carve-up (256B aligned)
    char* w = (char*)d_ws;
    auto take = [&](size_t bytes) {
        char* r = w;
        w += (bytes + 255) & ~(size_t)255;
        return r;
    };
    float*    nfd   = (float*)   take((size_t)NPIX * 3 * 4);        // 3 MB
    float*    bufA  = (float*)   take((size_t)NPIX * 64 * 4);       // 64 MB
    _Float16* bufP  = (_Float16*)take((size_t)NPIX * 64 * 2);       // 32 MB
    float*    res3  = (float*)   take((size_t)NPIX * 3 * 4);        // 3 MB
    _Float16* wpk1  = (_Float16*)take(4 * 32 * 16 * 2);             // 4 KB
    _Float16* wpk2  = (_Float16*)take(9 * 2 * 4 * 32 * 16 * 2);     // 72 KB
    float*    stats = (float*)   take(32 * 2 * 4);                  // reused GN1/GN2
    (void)ws_size; (void)in_sizes; (void)n_in; (void)out_size;

    // 1) depth -> normal_from_depth
    k_normal_from_depth<<<dim3(WW / 16, HH / 16, BATCH), dim3(16, 16), 0, stream>>>(
        initial_depth, x_fov, nfd);

    // 2) weight packing (independent of step 1)
    k_pack_w1<<<(2048 + 255) / 256, 256, 0, stream>>>(w1, wpk1);
    k_pack_w2<<<(36864 + 255) / 256, 256, 0, stream>>>(w2, wpk2);

    // 3) conv1 (WMMA, MT=4) -> relu -> bufA (NHWC f32)
    k_conv1_wmma<<<NPIX / 64, 128, 0, stream>>>(nfd, wpk1, b1, bufA);

    // 4) GN1 -> bufP (NHWC f16)
    k_gn_stats<<<32, 256, 0, stream>>>(bufA, stats);
    k_gn_apply_pack<<<(NPIX * 16) / 256, 256, 0, stream>>>(bufA, stats, g1, bt1, bufP);

    // 5) conv2 (WMMA, LDS-staged, 72 WMMAs/wave) -> relu -> bufA
    k_conv2_wmma<<<NPIX / 64, 128, 0, stream>>>(bufP, wpk2, b2, bufA);

    // 6) GN2 -> bufP
    k_gn_stats<<<32, 256, 0, stream>>>(bufA, stats);
    k_gn_apply_pack<<<(NPIX * 16) / 256, 256, 0, stream>>>(bufA, stats, g2, bt2, bufP);

    // 7) conv3 64->3
    k_conv3<<<NPIX / 256, 256, 0, stream>>>(bufP, w3, b3, res3);

    // 8) fused residual + normalize + concat + 1x1 conv + normalize + clip
    k_final<<<NPIX / 256, 256, 0, stream>>>(initial_normals, nfd, res3, wfc, bfc, out);
}
